// SpatialMSA_35424890257919
// MI455X (gfx1250) — matrix operand order repaired and checked by reading, not compiled, not run on hardware
//
#include <hip/hip_runtime.h>
#include <hip/hip_bf16.h>
#include <cmath>

// ---------------------------------------------------------------------------
// SpatialMSA for MI455X (gfx1250, wave32).
// All GEMMs run on v_wmma_f32_16x16x32_bf16 with f32 accumulation.
// Weights are converted once to bf16 *transposed* so B-fragments are two
// 16B loads along one row per lane; the 134MB bf16 weight set stays L2-resident.
// QKV projections are fused into one launch (grid.z selects q/k/v) so the
// projection phase runs 1536 waves instead of 3 x 512.
// ---------------------------------------------------------------------------

typedef __bf16 bf16;
typedef __attribute__((ext_vector_type(16))) __bf16 v16bf;
typedef __attribute__((ext_vector_type(8)))  __bf16 v8bf;
typedef __attribute__((ext_vector_type(8)))  float  v8f;

#define DEVFN static __device__ __forceinline__

// A-fragment (16x32 bf16, MxK): lane l<16 -> row l, K {k0..k0+7, k0+16..k0+23};
// lane l>=16 -> row l-16, K {k0+8..k0+15, k0+24..k0+31}.  Two 16B loads.
DEVFN v16bf load_frag_a(const bf16* base, int ld, int k0, int lane) {
  const bf16* p = base + (size_t)(lane & 15) * ld + (k0 + ((lane >> 4) << 3));
  v8bf lo = *(const v8bf*)p;
  v8bf hi = *(const v8bf*)(p + 16);
  v16bf r;
#pragma unroll
  for (int t = 0; t < 8; ++t) { r[t] = lo[t]; r[t + 8] = hi[t]; }
  return r;
}

// B-fragment (32x16 bf16, KxN) from B^T stored [N,K] row-major:
// lane l -> column n0+(l&15), K {k0+8h..k0+8h+7, k0+16+8h..k0+23+8h}, h=l>>4. Two 16B loads.
DEVFN v16bf load_frag_b(const bf16* base, int ld, int k0, int lane) {
  const bf16* p = base + (size_t)(lane & 15) * ld + (k0 + ((lane >> 4) << 3));
  v8bf lo = *(const v8bf*)p;
  v8bf hi = *(const v8bf*)(p + 16);
  v16bf r;
#pragma unroll
  for (int t = 0; t < 8; ++t) { r[t] = lo[t]; r[t + 8] = hi[t]; }
  return r;
}

DEVFN v8f wmma_bf16(v16bf a, v16bf b, v8f c) {
  return __builtin_amdgcn_wmma_f32_16x16x32_bf16(false, a, false, b, (short)0, c,
                                                 false, false);
}

// ---------------------------------------------------------------------------
// K0a: weight convert+transpose: W[k][n] f32 -> WT[n][k] bf16 (64x64 LDS tile)
// ---------------------------------------------------------------------------
__global__ __launch_bounds__(256) void wt_convert_kernel(
    const float* __restrict__ w0, const float* __restrict__ w1,
    const float* __restrict__ w2, const float* __restrict__ w3,
    bf16* __restrict__ o0, bf16* __restrict__ o1,
    bf16* __restrict__ o2, bf16* __restrict__ o3) {
  __shared__ float tile[64][65];
  const float* src; bf16* dst;
  switch (blockIdx.z) {
    case 0: src = w0; dst = o0; break;
    case 1: src = w1; dst = o1; break;
    case 2: src = w2; dst = o2; break;
    default: src = w3; dst = o3; break;
  }
  int k0 = blockIdx.y * 64, n0 = blockIdx.x * 64;
#pragma unroll
  for (int it = 0; it < 16; ++it) {
    int i = threadIdx.x + it * 256;
    int r = i >> 6, c = i & 63;
    tile[r][c] = src[(size_t)(k0 + r) * 4096 + n0 + c];
  }
  __syncthreads();
#pragma unroll
  for (int it = 0; it < 16; ++it) {
    int i = threadIdx.x + it * 256;
    int r = i >> 6, c = i & 63;
    dst[(size_t)(n0 + r) * 4096 + k0 + c] = (bf16)tile[c][r];
  }
}

// K0b: x f32 -> bf16 (natural [bc, dv] layout)
__global__ void x_convert_kernel(const float* __restrict__ xm,
                                 const float* __restrict__ xf,
                                 bf16* __restrict__ om, bf16* __restrict__ of,
                                 int n) {
  int i = blockIdx.x * blockDim.x + threadIdx.x;
  if (i < n) { om[i] = (bf16)xm[i]; of[i] = (bf16)xf[i]; }
}

// ---------------------------------------------------------------------------
// K1: fused QKV projections. blockIdx.z: 0 -> q = xm*Wq (f32 out),
// 1 -> k = xf*Wk (f32 out), 2 -> v = xf*Wv (bf16 out, natural layout).
// Wave tile 64x64 (16 v8f accumulators), WG tile 128x256 (8 waves, 2x4).
// ---------------------------------------------------------------------------
__global__ __launch_bounds__(256) void gemm_qkv_kernel(
    const bf16* __restrict__ xmb, const bf16* __restrict__ xfb,
    const bf16* __restrict__ WqT, const bf16* __restrict__ WkT,
    const bf16* __restrict__ WvT,
    float* __restrict__ qf, float* __restrict__ kf, bf16* __restrict__ vb,
    int K, int N) {
  const bf16* A; const bf16* BT;
  float* Cf = nullptr; bf16* Cb = nullptr;
  switch (blockIdx.z) {
    case 0: A = xmb; BT = WqT; Cf = qf; break;
    case 1: A = xfb; BT = WkT; Cf = kf; break;
    default: A = xfb; BT = WvT; Cb = vb; break;
  }
  int lane = threadIdx.x & 31, w = threadIdx.x >> 5;
  int m0 = blockIdx.y * 128 + (w >> 2) * 64;
  int n0 = blockIdx.x * 256 + (w & 3) * 64;
  v8f acc[4][4] = {};
  for (int k0 = 0; k0 < K; k0 += 32) {
    v16bf a[4], b[4];
#pragma unroll
    for (int i = 0; i < 4; ++i)
      a[i] = load_frag_a(A + (size_t)(m0 + 16 * i) * K, K, k0, lane);
#pragma unroll
    for (int j = 0; j < 4; ++j)
      b[j] = load_frag_b(BT + (size_t)(n0 + 16 * j) * K, K, k0, lane);
#pragma unroll
    for (int i = 0; i < 4; ++i)
#pragma unroll
      for (int j = 0; j < 4; ++j) acc[i][j] = wmma_bf16(a[i], b[j], acc[i][j]);
  }
  int cn = lane & 15, rb = (lane >> 4) << 3;
#pragma unroll
  for (int i = 0; i < 4; ++i)
#pragma unroll
    for (int j = 0; j < 4; ++j) {
      size_t col = (size_t)(n0 + 16 * j + cn);
#pragma unroll
      for (int r = 0; r < 8; ++r) {
        size_t row = (size_t)(m0 + 16 * i + rb + r);
        if (Cb) Cb[row * N + col] = (bf16)acc[i][j][r];
        else    Cf[row * N + col] = acc[i][j][r];
      }
    }
}

// ---------------------------------------------------------------------------
// K2: L2-norm over c (per batch, per column) + per-head transpose:
//   q[b,c,h*512+d] f32  ->  Qt[bh][d][c] bf16 (normalized over the 64 c's)
// blockIdx: (d-chunk of 128, bh, 0=q/1=k)
// ---------------------------------------------------------------------------
__global__ __launch_bounds__(256) void norm_t_kernel(
    const float* __restrict__ qf, const float* __restrict__ kf,
    bf16* __restrict__ Qt, bf16* __restrict__ Kt) {
  __shared__ float tile[64][129];
  __shared__ float inorm[128];
  const float* src = blockIdx.z ? kf : qf;
  bf16* dst = blockIdx.z ? Kt : Qt;
  int bh = blockIdx.y, b = bh >> 3, h = bh & 7;
  int d0 = blockIdx.x * 128;
#pragma unroll
  for (int it = 0; it < 32; ++it) {
    int i = threadIdx.x + it * 256;
    int c = i >> 7, dl = i & 127;
    tile[c][dl] = src[(size_t)(b * 64 + c) * 4096 + h * 512 + d0 + dl];
  }
  __syncthreads();
  if (threadIdx.x < 128) {
    float s = 0.f;
#pragma unroll
    for (int c = 0; c < 64; ++c) { float v = tile[c][threadIdx.x]; s += v * v; }
    inorm[threadIdx.x] = 1.0f / fmaxf(sqrtf(s), 1e-12f);
  }
  __syncthreads();
#pragma unroll
  for (int it = 0; it < 32; ++it) {
    int i = threadIdx.x + it * 256;
    int c = i & 63, dl = i >> 6;
    dst[(size_t)bh * 512 * 64 + (size_t)(d0 + dl) * 64 + c] =
        (bf16)(tile[c][dl] * inorm[dl]);
  }
}

// ---------------------------------------------------------------------------
// K3: fused attention per (b,h, 64-row d-block). 8 waves each own a 64-wide
// e-chunk of S = Kn*Qn^T (WMMA). Softmax stats via shfl_xor (16-lane halves,
// since acc rows split across lane halves) + LDS cross-wave. exp(S) staged
// to LDS in bf16 A-fragment layout, GEMM2 partial X = P*V (WMMA), reduced
// with LDS ds_add_f32 atomics, scaled by 1/rowsum, written transposed.
// ---------------------------------------------------------------------------
__global__ __launch_bounds__(256) void attn_kernel(
    const bf16* __restrict__ Qt, const bf16* __restrict__ Kt,
    const bf16* __restrict__ Vb, bf16* __restrict__ X2) {
  __shared__ __align__(16) bf16 Pl[8][64][40];  // per-wave P stage (16B-aligned rows)
  __shared__ float xred[64][65];
  __shared__ float maxp[8][64];
  __shared__ float sump[8][64];
  int lane = threadIdx.x & 31, w = threadIdx.x >> 5;
  int d0 = blockIdx.x * 64;
  int bh = blockIdx.y, b = bh >> 3, h = bh & 7;
  const bf16* Kh = Kt + (size_t)bh * 512 * 64;
  const bf16* Qh = Qt + (size_t)bh * 512 * 64;
  int e0 = w * 64;
  for (int i = threadIdx.x; i < 64 * 65; i += 256) (&xred[0][0])[i] = 0.f;

  // GEMM1: S[64 d x 64 e_chunk] = Kn[d,c] * Qn[e,c]^T, K=c=64
  v8f s[4][4] = {};
  for (int k0 = 0; k0 < 64; k0 += 32) {
    v16bf a[4], bb[4];
#pragma unroll
    for (int i = 0; i < 4; ++i)
      a[i] = load_frag_a(Kh + (size_t)(d0 + 16 * i) * 64, 64, k0, lane);
#pragma unroll
    for (int j = 0; j < 4; ++j)
      bb[j] = load_frag_b(Qh + (size_t)(e0 + 16 * j) * 64, 64, k0, lane);
#pragma unroll
    for (int i = 0; i < 4; ++i)
#pragma unroll
      for (int j = 0; j < 4; ++j) s[i][j] = wmma_bf16(a[i], bb[j], s[i][j]);
  }
  int half8 = (lane >> 4) << 3;

  // row max: per-lane over j, then across the 16-lane half, then across waves
  float pm[4][8];
#pragma unroll
  for (int i = 0; i < 4; ++i)
#pragma unroll
    for (int r = 0; r < 8; ++r) {
      float m = s[i][0][r];
#pragma unroll
      for (int j = 1; j < 4; ++j) m = fmaxf(m, s[i][j][r]);
      pm[i][r] = m;
    }
#pragma unroll
  for (int mask = 1; mask <= 8; mask <<= 1)
#pragma unroll
    for (int i = 0; i < 4; ++i)
#pragma unroll
      for (int r = 0; r < 8; ++r)
        pm[i][r] = fmaxf(pm[i][r], __shfl_xor(pm[i][r], mask, 32));
  if ((lane & 15) == 0)
#pragma unroll
    for (int i = 0; i < 4; ++i)
#pragma unroll
      for (int r = 0; r < 8; ++r) maxp[w][16 * i + half8 + r] = pm[i][r];
  __syncthreads();
  float gm[4][8];
#pragma unroll
  for (int i = 0; i < 4; ++i)
#pragma unroll
    for (int r = 0; r < 8; ++r) {
      float m = maxp[0][16 * i + half8 + r];
#pragma unroll
      for (int wv = 1; wv < 8; ++wv) m = fmaxf(m, maxp[wv][16 * i + half8 + r]);
      gm[i][r] = m;
    }
  // exp + row sums (S is overwritten with P = exp(S - max))
  float ps[4][8] = {};
#pragma unroll
  for (int i = 0; i < 4; ++i)
#pragma unroll
    for (int j = 0; j < 4; ++j)
#pragma unroll
      for (int r = 0; r < 8; ++r) {
        float p = __expf(s[i][j][r] - gm[i][r]);
        s[i][j][r] = p;
        ps[i][r] += p;
      }
#pragma unroll
  for (int mask = 1; mask <= 8; mask <<= 1)
#pragma unroll
    for (int i = 0; i < 4; ++i)
#pragma unroll
      for (int r = 0; r < 8; ++r) ps[i][r] += __shfl_xor(ps[i][r], mask, 32);
  if ((lane & 15) == 0)
#pragma unroll
    for (int i = 0; i < 4; ++i)
#pragma unroll
      for (int r = 0; r < 8; ++r) sump[w][16 * i + half8 + r] = ps[i][r];

  // GEMM2: partial X[64 d x 64 c] over this wave's e-chunk, 2 rounds of K=32.
  // B^T = V^T[c, e] == v natural layout [b*64+c][h*512+e], contiguous in e.
  const bf16* VTb = Vb + (size_t)(b * 64) * 4096 + h * 512 + e0;
  v8f xa[4][4] = {};
  for (int rd = 0; rd < 2; ++rd) {
#pragma unroll
    for (int i = 0; i < 4; ++i)
#pragma unroll
      for (int jj = 0; jj < 2; ++jj)
#pragma unroll
        for (int r = 0; r < 8; ++r)
          Pl[w][16 * i + half8 + r][16 * jj + (lane & 15)] =
              (bf16)s[i][2 * rd + jj][r];
    __syncthreads();
    v16bf a[4], bb[4];
#pragma unroll
    for (int i = 0; i < 4; ++i)
      a[i] = load_frag_a(&Pl[w][16 * i][0], 40, 0, lane);
#pragma unroll
    for (int jn = 0; jn < 4; ++jn)
      bb[jn] = load_frag_b(VTb + (size_t)(16 * jn) * 4096 + 32 * rd, 4096, 0, lane);
#pragma unroll
    for (int i = 0; i < 4; ++i)
#pragma unroll
      for (int jn = 0; jn < 4; ++jn) xa[i][jn] = wmma_bf16(a[i], bb[jn], xa[i][jn]);
    __syncthreads();
  }
  // reduce partial X across waves (ds_add_f32 atomics)
#pragma unroll
  for (int i = 0; i < 4; ++i)
#pragma unroll
    for (int jn = 0; jn < 4; ++jn)
#pragma unroll
      for (int r = 0; r < 8; ++r)
        atomicAdd(&xred[16 * i + half8 + r][16 * jn + (lane & 15)], xa[i][jn][r]);
  __syncthreads();
  // scale by 1/rowsum, write transposed into x2[bc, hd]
  for (int i = threadIdx.x; i < 4096; i += 256) {
    int dl = i & 63, c = i >> 6;
    float gs = 0.f;
#pragma unroll
    for (int wv = 0; wv < 8; ++wv) gs += sump[wv][dl];
    float val = xred[dl][c] / gs;
    X2[(size_t)(b * 64 + c) * 4096 + h * 512 + d0 + dl] = (bf16)val;
  }
}

// ---------------------------------------------------------------------------
// K4: out = gelu(x2 @ Wp + bp), exact erf GELU, f32 output.
// ---------------------------------------------------------------------------
__global__ __launch_bounds__(256) void gemm_gelu_kernel(
    const bf16* __restrict__ A, const bf16* __restrict__ BT,
    const float* __restrict__ bias, float* __restrict__ out, int K, int N) {
  int lane = threadIdx.x & 31, w = threadIdx.x >> 5;
  int m0 = blockIdx.y * 128 + (w >> 2) * 64;
  int n0 = blockIdx.x * 256 + (w & 3) * 64;
  v8f acc[4][4] = {};
  for (int k0 = 0; k0 < K; k0 += 32) {
    v16bf a[4], b[4];
#pragma unroll
    for (int i = 0; i < 4; ++i)
      a[i] = load_frag_a(A + (size_t)(m0 + 16 * i) * K, K, k0, lane);
#pragma unroll
    for (int j = 0; j < 4; ++j)
      b[j] = load_frag_b(BT + (size_t)(n0 + 16 * j) * K, K, k0, lane);
#pragma unroll
    for (int i = 0; i < 4; ++i)
#pragma unroll
      for (int j = 0; j < 4; ++j) acc[i][j] = wmma_bf16(a[i], b[j], acc[i][j]);
  }
  int cn = lane & 15, rb = (lane >> 4) << 3;
#pragma unroll
  for (int i = 0; i < 4; ++i)
#pragma unroll
    for (int j = 0; j < 4; ++j) {
      size_t col = (size_t)(n0 + 16 * j + cn);
      float bv = bias[col];
#pragma unroll
      for (int r = 0; r < 8; ++r) {
        float v = acc[i][j][r] + bv;
        float g = 0.5f * v * (1.0f + erff(v * 0.7071067811865475f));
        out[(size_t)(m0 + 16 * i + rb + r) * N + col] = g;
      }
    }
}

// ---------------------------------------------------------------------------
extern "C" void kernel_launch(void* const* d_in, const int* in_sizes, int n_in,
                              void* d_out, int out_size, void* d_ws, size_t ws_size,
                              hipStream_t stream) {
  const float* x_msi = (const float*)d_in[0];
  const float* x_fu  = (const float*)d_in[1];
  const float* Wq = (const float*)d_in[2];
  const float* Wk = (const float*)d_in[3];
  const float* Wv = (const float*)d_in[4];
  const float* Wp = (const float*)d_in[5];
  const float* bp = (const float*)d_in[6];
  float* out = (float*)d_out;

  const size_t WSZ = (size_t)4096 * 4096;  // weight elements
  const size_t XSZ = (size_t)512 * 4096;   // activation elements
  char* p = (char*)d_ws;
  bf16* WqT = (bf16*)p; p += WSZ * 2;
  bf16* WkT = (bf16*)p; p += WSZ * 2;
  bf16* WvT = (bf16*)p; p += WSZ * 2;
  bf16* WpT = (bf16*)p; p += WSZ * 2;
  bf16* xmb = (bf16*)p; p += XSZ * 2;
  bf16* xfb = (bf16*)p; p += XSZ * 2;
  float* qf = (float*)p; p += XSZ * 4;
  float* kf = (float*)p; p += XSZ * 4;
  bf16* vb  = (bf16*)p; p += XSZ * 2;
  bf16* Qt  = (bf16*)p; p += XSZ * 2;
  bf16* Kt  = (bf16*)p; p += XSZ * 2;
  bf16* x2  = (bf16*)p; p += XSZ * 2;

  wt_convert_kernel<<<dim3(64, 64, 4), 256, 0, stream>>>(Wq, Wk, Wv, Wp,
                                                         WqT, WkT, WvT, WpT);
  x_convert_kernel<<<dim3(8192), 256, 0, stream>>>(x_msi, x_fu, xmb, xfb, (int)XSZ);
  gemm_qkv_kernel<<<dim3(16, 4, 3), 256, 0, stream>>>(xmb, xfb, WqT, WkT, WvT,
                                                      qf, kf, vb, 4096, 4096);
  norm_t_kernel<<<dim3(4, 64, 2), 256, 0, stream>>>(qf, kf, Qt, Kt);
  attn_kernel<<<dim3(8, 64), 256, 0, stream>>>(Qt, Kt, vb, x2);
  gemm_gelu_kernel<<<dim3(16, 4), 256, 0, stream>>>(x2, WpT, bp, out, 4096, 4096);
}
